// MultiMeanDisplacerNet_8804682957404
// MI455X (gfx1250) — matrix operand order repaired and checked by reading, not compiled, not run on hardware
//
#include <hip/hip_runtime.h>

#define N_V 8192
#define CT  256
#define NEG 0.2f

typedef __attribute__((ext_vector_type(16))) __bf16 v16bf;
typedef __attribute__((ext_vector_type(8)))  float  v8f;
typedef unsigned uint32x4_t __attribute__((ext_vector_type(4)));
typedef int      int32x8_t  __attribute__((ext_vector_type(8)));
typedef int      int32x4_t  __attribute__((ext_vector_type(4)));

// ---------- bf16 helpers (bit-level) ----------
__device__ inline unsigned short f2bf(float f){
  unsigned u = __float_as_uint(f);
  unsigned r = 0x7FFFu + ((u >> 16) & 1u);   // round-to-nearest-even
  return (unsigned short)((u + r) >> 16);
}
__device__ inline float bf2f(unsigned short h){
  return __uint_as_float(((unsigned)h) << 16);
}

// ---------- WMMA fragment loaders (layouts per CDNA5 ISA 7.12.2) ----------
// A 16x32 bf16: lane L<16 -> row m0+L, K = k0+{0..7, 16..23}
//               lane L>=16 -> row m0+L-16, K = k0+{8..15, 24..31}
__device__ inline v16bf load_frag_a(const unsigned short* A, int ldk, int m0, int k0, int lane){
  union { uint4 u[2]; v16bf v; } f;
  int half = lane >> 4;
  const unsigned short* p = A + (size_t)(m0 + (lane & 15)) * ldk + (k0 + half * 8);
  f.u[0] = *(const uint4*)(p);
  f.u[1] = *(const uint4*)(p + 16);
  return f.v;
}
// B 32x16 bf16 from Bt (row-major [Ncols][K]):
// lane L<16 -> col n0+L, K = k0+{0..7, 16..23}; lane L>=16 -> col n0+L-16, K = k0+{8..15, 24..31}
__device__ inline v16bf load_frag_b(const unsigned short* Bt, int ldk, int n0, int k0, int lane){
  union { uint4 u[2]; v16bf v; } f;
  int half = lane >> 4;
  const unsigned short* p = Bt + (size_t)(n0 + (lane & 15)) * ldk + (k0 + half * 8);
  f.u[0] = *(const uint4*)(p);
  f.u[1] = *(const uint4*)(p + 16);
  return f.v;
}

// ---------- Tensor Data Mover: 2D tile (rows x cols bf16) global -> LDS ----------
// D# per CDNA5 ISA ch.8: group0 {count|lds_addr|global_addr|type=2},
// group1 {data_size=2B, tensor_dim0=cols, tensor_dim1=rows, tile_dim0=cols,
//          tile_dim1=rows, dim0_stride=stride}; remaining groups zero (2D tile).
__device__ inline void tdm_load_2d(const unsigned short* gsrc, unsigned lds_addr,
                                   int cols, int rows, int row_stride){
  unsigned long long ga = (unsigned long long)(uintptr_t)gsrc;
  uint32x4_t g0;
  g0[0] = 1u;                                  // count=1, user descriptor
  g0[1] = lds_addr;                            // lds_addr (bytes)
  g0[2] = (unsigned)ga;                        // global_addr[31:0]
  g0[3] = (unsigned)(ga >> 32) | (2u << 30);   // global_addr[56:32] | type=2
  int32x8_t g1;
  g1[0] = 1 << 16;                             // wg_mask=0 | data_size=1 (2 bytes)
  g1[1] = (cols & 0xFFFF) << 16;               // atomic_addr=0 | tensor_dim0.lo
  g1[2] = ((unsigned)cols >> 16) | ((rows & 0xFFFF) << 16); // dim0.hi | tensor_dim1.lo
  g1[3] = (cols & 0xFFFF) << 16;               // tensor_dim1.hi=0 | tile_dim0
  g1[4] = (rows & 0xFFFF);                     // tile_dim1 | tile_dim2=0
  g1[5] = row_stride;                          // tensor_dim0_stride[31:0]
  g1[6] = 0;
  g1[7] = 0;
  int32x4_t z4 = {0, 0, 0, 0};
  int32x8_t z8 = {0, 0, 0, 0, 0, 0, 0, 0};
  __builtin_amdgcn_tensor_load_to_lds(g0, g1, z4, z4, z8, 0);
}
__device__ inline unsigned lds_off(const void* p){
  return (unsigned)(uintptr_t)p;               // flat->LDS uses addr[31:0] (ISA 10.2)
}

// ---------- generic WMMA GEMM: C[M,Nc] f32 = A[M,K]bf16 x Bt[Nc,K]bf16 ----------
// Each wave computes a 16x64 block (A fragment reused across 4 N-tiles).
__global__ void k_wmma_gemm(const unsigned short* __restrict__ A,
                            const unsigned short* __restrict__ Bt,
                            float* __restrict__ C, const float* __restrict__ bias,
                            int M, int Nc, int K, int act){
  int lane = threadIdx.x & 31;
  int n0 = blockIdx.x * 64;
  int m0 = blockIdx.y * 16;
  v8f acc[4] = {{}, {}, {}, {}};
  for (int k0 = 0; k0 < K; k0 += 32){
    v16bf a = load_frag_a(A, K, m0, k0, lane);
    #pragma unroll
    for (int j = 0; j < 4; j++){
      v16bf b = load_frag_b(Bt, K, n0 + 16 * j, k0, lane);
      acc[j] = __builtin_amdgcn_wmma_f32_16x16x32_bf16(false, a, false, b, (short)0, acc[j], false, false);
    }
  }
  int half = lane >> 4;
  #pragma unroll
  for (int j = 0; j < 4; j++){
    int col = n0 + 16 * j + (lane & 15);
    float bv = bias ? bias[col] : 0.f;
    #pragma unroll
    for (int r = 0; r < 8; r++){
      int row = m0 + r + half * 8;
      float v = acc[j][r] + bv;
      if (act) v = v > 0.f ? v : 0.f;
      C[(size_t)row * Nc + col] = v;
    }
  }
}

// ---------- fused kNN: TDM-staged panels + WMMA + running top-16 ----------
// Block = 4 waves. Wave w owns rows [m0, m0+16). A fragments live in registers
// for the whole sweep; the shared 16-row B panel is double-buffered in LDS via
// the Tensor Data Mover (TENSORcnt), issued by wave 0.
template <int C>
__global__ void k_knn(const unsigned short* __restrict__ X, const float* __restrict__ sq,
                      int* __restrict__ idx){
  __shared__ unsigned short panel[2][16 * C];
  __shared__ float smem[4][16][16];
  const int KF = C / 32;
  int lane = threadIdx.x & 31;
  int wv   = threadIdx.x >> 5;
  int m0   = blockIdx.x * 64 + wv * 16;
  int half = lane >> 4;

  if (wv == 0) tdm_load_2d(X, lds_off(&panel[0][0]), C, 16, C);

  v16bf afr[KF];
  #pragma unroll
  for (int kf = 0; kf < KF; kf++) afr[kf] = load_frag_a(X, C, m0, kf * 32, lane);

  float bd[16]; int bi[16];
  #pragma unroll
  for (int t = 0; t < 16; t++){ bd[t] = 1e30f; bi[t] = 0; }
  float worst = 1e30f; int wpos = 0;

  const int NT = N_V / 16;
  for (int i = 0; i < NT; i++){
    int n0 = i * 16;
    if (wv == 0){
      if (i + 1 < NT){
        tdm_load_2d(X + (size_t)(n0 + 16) * C, lds_off(&panel[(i + 1) & 1][0]), C, 16, C);
        __builtin_amdgcn_s_wait_tensorcnt(1);   // current panel complete
      } else {
        __builtin_amdgcn_s_wait_tensorcnt(0);
      }
    }
    __syncthreads();                            // panel[i&1] visible to all waves

    v8f acc = {};
    #pragma unroll
    for (int kf = 0; kf < KF; kf++){
      v16bf b = load_frag_b(&panel[i & 1][0], C, 0, kf * 32, lane);
      acc = __builtin_amdgcn_wmma_f32_16x16x32_bf16(false, afr[kf], false, b, (short)0, acc, false, false);
    }
    int   n   = n0 + (lane & 15);
    float sqn = sq[n];
    #pragma unroll
    for (int r = 0; r < 8; r++){
      int m = m0 + r + half * 8;
      float s = sqn - 2.f * acc[r];             // sq[m] row-constant: drops out of top-k
      if (n == m) s = 1e30f;                    // exclude self
      smem[wv][r + half * 8][lane & 15] = s;
    }
    __syncthreads();
    if (lane < 16){
      for (int j = 0; j < 16; j++){
        float s = smem[wv][lane][j];
        if (s < worst){
          int nn = n0 + j;
          #pragma unroll
          for (int t = 0; t < 16; t++) if (t == wpos){ bd[t] = s; bi[t] = nn; }
          worst = -1e30f;
          #pragma unroll
          for (int t = 0; t < 16; t++) if (bd[t] > worst){ worst = bd[t]; wpos = t; }
        }
      }
    }
    __syncthreads();
  }
  if (lane < 16){
    int m = m0 + lane;
    #pragma unroll
    for (int t = 0; t < 16; t++) idx[m * 16 + t] = bi[t];
  }
}

// ---------- GATv2 attention + aggregation (f32 VALU, deterministic reduce) ----------
__global__ void k_attn(const float* __restrict__ xl, const float* __restrict__ xr,
                       const int* __restrict__ idx, const float* __restrict__ att,
                       const float* __restrict__ bias, float* __restrict__ y, int C){
  __shared__ float red[16][8];
  int n = blockIdx.x;
  int tid = threadIdx.x;
  int lane = tid & 31, wv = tid >> 5;
  int nb[16];
  #pragma unroll
  for (int k = 0; k < 16; k++) nb[k] = idx[n * 16 + k];
  float ep[16];
  #pragma unroll
  for (int k = 0; k < 16; k++) ep[k] = 0.f;
  for (int c = tid; c < C; c += 256){
    float xrv = xr[(size_t)n * C + c];
    float av  = att[c];
    #pragma unroll
    for (int k = 0; k < 16; k++){
      float v = xl[(size_t)nb[k] * C + c] + xrv;
      v = v > 0.f ? v : NEG * v;
      ep[k] += av * v;
    }
  }
  #pragma unroll
  for (int k = 0; k < 16; k++){
    float v = ep[k];
    for (int o = 16; o > 0; o >>= 1) v += __shfl_xor(v, o, 32);
    if (lane == 0) red[k][wv] = v;
  }
  __syncthreads();
  float e[16], mx = -1e30f;
  #pragma unroll
  for (int k = 0; k < 16; k++){
    float s = 0.f;
    #pragma unroll
    for (int w = 0; w < 8; w++) s += red[k][w];
    e[k] = s; mx = fmaxf(mx, s);
  }
  float den = 0.f, wgt[16];
  #pragma unroll
  for (int k = 0; k < 16; k++){ wgt[k] = __expf(e[k] - mx); den += wgt[k]; }
  float inv = 1.f / den;
  for (int c = tid; c < C; c += 256){
    float a = bias[c];
    #pragma unroll
    for (int k = 0; k < 16; k++) a += wgt[k] * inv * xl[(size_t)nb[k] * C + c];
    y[(size_t)n * C + c] = fmaxf(a, 0.f);
  }
}

// ---------- feature transform: 3 masked tiny GEMMs (K=3/6/3) ----------
__global__ void k_ft(const float* __restrict__ x,
                     const float* __restrict__ W0, const float* __restrict__ b0,
                     const float* __restrict__ W1, const float* __restrict__ b1,
                     const float* __restrict__ W2, const float* __restrict__ b2,
                     unsigned short* __restrict__ xb){
  int t = blockIdx.x * blockDim.x + threadIdx.x;
  if (t >= 3 * N_V * CT) return;
  int c = t % CT;
  int n = (t / CT) % N_V;
  int b = t / (CT * N_V);
  const float* xrow = x + n * 12;
  float v;
  if (b == 0)
    v = b0[c] + xrow[0]*W0[c] + xrow[1]*W0[CT+c] + xrow[2]*W0[2*CT+c];
  else if (b == 1)
    v = b1[c] + xrow[3]*W1[c] + xrow[4]*W1[CT+c] + xrow[5]*W1[2*CT+c]
              + xrow[6]*W1[3*CT+c] + xrow[7]*W1[4*CT+c] + xrow[8]*W1[5*CT+c];
  else
    v = b2[c] + xrow[9]*W2[c] + xrow[10]*W2[CT+c] + xrow[11]*W2[2*CT+c];
  xb[t] = f2bf(v);
}

__global__ void k_sqnorm(const unsigned short* __restrict__ X, float* __restrict__ sq, int C){
  int n = blockIdx.x * blockDim.x + threadIdx.x;
  if (n >= N_V) return;
  float s = 0.f;
  for (int c = 0; c < C; c++){ float v = bf2f(X[(size_t)n * C + c]); s += v * v; }
  sq[n] = s;
}

__global__ void k_transpose_bf16(const float* __restrict__ W, unsigned short* __restrict__ Wt,
                                 int R, int Cc){
  int t = blockIdx.x * blockDim.x + threadIdx.x;
  if (t >= R * Cc) return;
  int r = t / Cc, c = t % Cc;
  Wt[(size_t)c * R + r] = f2bf(W[(size_t)r * Cc + c]);
}

__global__ void k_mean3_post(const float* __restrict__ y0, const float* __restrict__ y1,
                             const float* __restrict__ y2, unsigned short* __restrict__ xnext,
                             unsigned short* __restrict__ cat, int C, int catld, int cofs){
  int t = blockIdx.x * blockDim.x + threadIdx.x;
  if (t >= N_V * C) return;
  int n = t / C, c = t % C;
  unsigned short h = f2bf((y0[t] + y1[t] + y2[t]) * (1.f / 3.f));
  xnext[t] = h;
  cat[(size_t)n * catld + cofs + c] = h;
}

__global__ void k_post(const float* __restrict__ y, unsigned short* __restrict__ xnext,
                       unsigned short* __restrict__ cat, int C, int catld, int cofs){
  int t = blockIdx.x * blockDim.x + threadIdx.x;
  if (t >= N_V * C) return;
  int n = t / C, c = t % C;
  unsigned short h = f2bf(y[t]);
  if (xnext) xnext[t] = h;
  cat[(size_t)n * catld + cofs + c] = h;
}

__global__ void k_f32_to_bf16(const float* __restrict__ s, unsigned short* __restrict__ d, int n){
  int t = blockIdx.x * blockDim.x + threadIdx.x;
  if (t < n) d[t] = f2bf(s[t]);
}

__global__ void k_mlp_out(const float* __restrict__ h, const float* __restrict__ W,
                          const float* __restrict__ b, float* __restrict__ out){
  int t = blockIdx.x * blockDim.x + threadIdx.x;
  if (t >= N_V * 3) return;
  int n = t / 3, j = t % 3;
  float s = b[j];
  for (int c = 0; c < 64; c++) s += h[n * 64 + c] * W[c * 3 + j];
  out[t] = s;
}

extern "C" void kernel_launch(void* const* d_in, const int* in_sizes, int n_in,
                              void* d_out, int out_size, void* d_ws, size_t ws_size,
                              hipStream_t stream){
  (void)in_sizes; (void)n_in; (void)out_size; (void)ws_size;
  // input order: x, ft[0..2]{W,b}, gat[0..3]{Wl,Wr,att,b}, mlp[0..2]{W,b}
  const float* x = (const float*)d_in[0];
  const float* ftW[3] = {(const float*)d_in[1], (const float*)d_in[3], (const float*)d_in[5]};
  const float* ftB[3] = {(const float*)d_in[2], (const float*)d_in[4], (const float*)d_in[6]};
  const float *Wl[4], *Wr[4], *att[4], *gb[4];
  for (int l = 0; l < 4; l++){
    Wl[l]  = (const float*)d_in[7 + l*4 + 0];
    Wr[l]  = (const float*)d_in[7 + l*4 + 1];
    att[l] = (const float*)d_in[7 + l*4 + 2];
    gb[l]  = (const float*)d_in[7 + l*4 + 3];
  }
  const float* mW[3] = {(const float*)d_in[23], (const float*)d_in[25], (const float*)d_in[27]};
  const float* mB[3] = {(const float*)d_in[24], (const float*)d_in[26], (const float*)d_in[28]};

  // ---- workspace suballocation (256B aligned) ----
  char* base = (char*)d_ws;
  size_t off = 0;
  auto alloc = [&](size_t bytes)->void*{
    void* p = base + off;
    off += bytes;
    off = (off + 255) & ~(size_t)255;
    return p;
  };
  unsigned short* xb   = (unsigned short*)alloc((size_t)3 * N_V * CT * 2);
  unsigned short* xcur = (unsigned short*)alloc((size_t)N_V * 512 * 2);
  unsigned short* catb = (unsigned short*)alloc((size_t)N_V * 1792 * 2);
  float* sq  = (float*)alloc((size_t)N_V * 4);
  int*   idx = (int*)  alloc((size_t)N_V * 16 * 4);
  float* xl  = (float*)alloc((size_t)N_V * 512 * 4);
  float* xr  = (float*)alloc((size_t)N_V * 512 * 4);
  float* y0  = (float*)alloc((size_t)N_V * 512 * 4);
  float* y1  = (float*)alloc((size_t)N_V * CT * 4);
  float* y2  = (float*)alloc((size_t)N_V * CT * 4);
  float* h1  = (float*)alloc((size_t)N_V * 256 * 4);
  unsigned short* h1b = (unsigned short*)alloc((size_t)N_V * 256 * 2);
  float* h2  = (float*)alloc((size_t)N_V * 64 * 4);
  int INC[4] = {256, 256, 512, 512}, OUC[4] = {256, 512, 512, 512};
  unsigned short *Wlt[4], *Wrt[4];
  for (int l = 0; l < 4; l++){
    Wlt[l] = (unsigned short*)alloc((size_t)INC[l] * OUC[l] * 2);
    Wrt[l] = (unsigned short*)alloc((size_t)INC[l] * OUC[l] * 2);
  }
  unsigned short* W0t = (unsigned short*)alloc((size_t)1792 * 256 * 2);
  unsigned short* W1t = (unsigned short*)alloc((size_t)256 * 64 * 2);

  auto tlaunch = [&](const float* W, unsigned short* Wt, int R, int Cc){
    int tot = R * Cc;
    k_transpose_bf16<<<(tot + 255) / 256, 256, 0, stream>>>(W, Wt, R, Cc);
  };
  for (int l = 0; l < 4; l++){
    tlaunch(Wl[l], Wlt[l], INC[l], OUC[l]);
    tlaunch(Wr[l], Wrt[l], INC[l], OUC[l]);
  }
  tlaunch(mW[0], W0t, 1792, 256);
  tlaunch(mW[1], W1t, 256, 64);

  auto gemm = [&](const unsigned short* A, const unsigned short* Bt, float* C,
                  const float* bias, int M, int Nc, int K, int act){
    dim3 g(Nc / 64, M / 16);
    k_wmma_gemm<<<g, 32, 0, stream>>>(A, Bt, C, bias, M, Nc, K, act);
  };
  auto knn = [&](const unsigned short* X, int Ci){
    k_sqnorm<<<(N_V + 255) / 256, 256, 0, stream>>>(X, sq, Ci);
    if (Ci == 256) k_knn<256><<<N_V / 64, 128, 0, stream>>>(X, sq, idx);
    else           k_knn<512><<<N_V / 64, 128, 0, stream>>>(X, sq, idx);
  };

  // ---- feature transform -> xb (bf16, [3][N][256]) ----
  k_ft<<<(3 * N_V * CT + 255) / 256, 256, 0, stream>>>(
      x, ftW[0], ftB[0], ftW[1], ftB[1], ftW[2], ftB[2], xb);

  // ---- layer 1: 3 graphs, mean over batches ----
  float* ys[3] = {y0, y1, y2};
  for (int b = 0; b < 3; b++){
    const unsigned short* xbb = xb + (size_t)b * N_V * CT;
    knn(xbb, 256);
    gemm(xbb, Wlt[0], xl, nullptr, N_V, 256, 256, 0);
    gemm(xbb, Wrt[0], xr, nullptr, N_V, 256, 256, 0);
    k_attn<<<N_V, 256, 0, stream>>>(xl, xr, idx, att[0], gb[0], ys[b], 256);
  }
  k_mean3_post<<<(N_V * 256 + 255) / 256, 256, 0, stream>>>(y0, y1, y2, xcur, catb, 256, 1792, 0);

  // ---- layers 2..4 ----
  int cofs = 256;
  for (int l = 1; l < 4; l++){
    int Ci = INC[l], Co = OUC[l];
    knn(xcur, Ci);
    gemm(xcur, Wlt[l], xl, nullptr, N_V, Co, Ci, 0);
    gemm(xcur, Wrt[l], xr, nullptr, N_V, Co, Ci, 0);
    k_attn<<<N_V, 256, 0, stream>>>(xl, xr, idx, att[l], gb[l], y0, Co);
    k_post<<<(N_V * Co + 255) / 256, 256, 0, stream>>>(
        y0, (l < 3) ? xcur : (unsigned short*)nullptr, catb, Co, 1792, cofs);
    cofs += Co;
  }

  // ---- MLP head ----
  gemm(catb, W0t, h1, mB[0], N_V, 256, 1792, 1);
  k_f32_to_bf16<<<(N_V * 256 + 255) / 256, 256, 0, stream>>>(h1, h1b, N_V * 256);
  gemm(h1b, W1t, h2, mB[1], N_V, 64, 256, 1);
  k_mlp_out<<<(N_V * 3 + 255) / 256, 256, 0, stream>>>(h2, mW[2], mB[2], (float*)d_out);
}
